// FullConvRes1_37434934952365
// MI455X (gfx1250) — matrix operand order repaired and checked by reading, not compiled, not run on hardware
//
#include <hip/hip_runtime.h>

typedef __attribute__((ext_vector_type(16))) _Float16 v16h;
typedef __attribute__((ext_vector_type(8)))  float    v8f;

// Problem constants (B,C,H,W)=(4,64,64,64), C8=8, OC=64, K9=9, HW=4096
// Conv-GEMM K-dim: kappa = k*64 + c, 0..575 = 18 WMMA steps of 32 (pairs share k).

// ---------------- Kernel A: q/k 1x1 conv projections -> (B,8,4096) ----------
__global__ __launch_bounds__(256) void qk_proj_kernel(
    const float* __restrict__ x,
    const float* __restrict__ qw, const float* __restrict__ qbias,
    const float* __restrict__ kw, const float* __restrict__ kbias,
    float* __restrict__ qo, float* __restrict__ ko)
{
    int g = blockIdx.x * 256 + threadIdx.x;   // 0..16383 = b*4096+n
    int b = g >> 12;
    int n = g & 4095;
    float qa[8], ka[8];
#pragma unroll
    for (int d = 0; d < 8; ++d) { qa[d] = qbias[d]; ka[d] = kbias[d]; }
    const float* xp = x + ((size_t)b << 18) + n;        // x[b, c, n], stride 4096 over c
    for (int c = 0; c < 64; ++c) {
        float xv = xp[(size_t)c << 12];
#pragma unroll
        for (int d = 0; d < 8; ++d) {
            qa[d] = fmaf(qw[(d << 6) + c], xv, qa[d]); // qw uniform -> scalar loads
            ka[d] = fmaf(kw[(d << 6) + c], xv, ka[d]);
        }
    }
#pragma unroll
    for (int d = 0; d < 8; ++d) {
        qo[(((b << 3) + d) << 12) + n] = qa[d];
        ko[(((b << 3) + d) << 12) + n] = ka[d];
    }
}

// ---------------- Kernel B: fused energy row + top-9 selection --------------
__device__ __forceinline__ void insert9(float v, int s, float (&tv)[9], int (&ti)[9])
{
    if (v <= tv[8]) return;                    // strict: earlier index wins ties
    tv[8] = v; ti[8] = s;
#pragma unroll
    for (int p = 8; p > 0; --p) {
        if (tv[p] > tv[p - 1]) {
            float fv = tv[p]; tv[p] = tv[p - 1]; tv[p - 1] = fv;
            int   fi = ti[p]; ti[p] = ti[p - 1]; ti[p - 1] = fi;
        }
    }
}

__global__ __launch_bounds__(32) void topk9_kernel(
    const float* __restrict__ q, const float* __restrict__ k,
    int* __restrict__ idxo)
{
    int wid  = blockIdx.x;                    // 0..16383: one wave per (b,t)
    int b    = wid >> 12;
    int trow = wid & 4095;
    int lane = threadIdx.x;

    float kv[8];
#pragma unroll
    for (int d = 0; d < 8; ++d) kv[d] = k[(((b << 3) + d) << 12) + trow];

    float tv[9]; int ti[9];
#pragma unroll
    for (int p = 0; p < 9; ++p) { tv[p] = -__builtin_inff(); ti[p] = 0; }

    const float* qb = q + ((size_t)b << 15);  // b*8*4096
    for (int i = 0; i < 128; ++i) {           // each lane scans 128 s positions
        int s = lane + (i << 5);
        float e = 0.f;
#pragma unroll
        for (int d = 0; d < 8; ++d) e = fmaf(kv[d], qb[(d << 12) + s], e);
        insert9(e, s, tv, ti);
    }

    // log2(32) sorted-list merge across lanes
#pragma unroll
    for (int off = 16; off >= 1; off >>= 1) {
        float pv[9]; int pi[9];
#pragma unroll
        for (int p = 0; p < 9; ++p) {
            pv[p] = __shfl_xor(tv[p], off, 32);
            pi[p] = __shfl_xor(ti[p], off, 32);
        }
#pragma unroll
        for (int p = 0; p < 9; ++p) insert9(pv[p], pi[p], tv, ti);
    }

    if (lane == 0) {
        // sort the 9 winning indices ascending (reference sorts idx)
#pragma unroll
        for (int a = 0; a < 9; ++a)
#pragma unroll
            for (int c = 0; c < 8; ++c)
                if (ti[c + 1] < ti[c]) { int f = ti[c]; ti[c] = ti[c + 1]; ti[c + 1] = f; }
        int base = ((b << 12) + trow) * 9;
#pragma unroll
        for (int p = 0; p < 9; ++p) idxo[base + p] = ti[p];
    }
}

// ------------- Kernel P1: pre-pack conv weight into WMMA A-fragment layout --
// A is 16x32 f16 per tile: lane = M (mod 16); element e of v16h holds
// K_local = (lane>>4)*16 + e   (contiguous order, same as B).
// Storage: wA[((t*4+ot)*32 + lane)*16 + e], t = K step (0..17), ot = M tile.
__global__ __launch_bounds__(256) void wprep_kernel(
    const float* __restrict__ w, _Float16* __restrict__ wA)
{
    int tid = blockIdx.x * 256 + threadIdx.x;
    if (tid >= 18 * 4 * 32 * 16) return;
    int e    = tid & 15;
    int lane = (tid >> 4) & 31;
    int ot   = (tid >> 9) & 3;
    int t    = tid >> 11;
    int hi   = lane >> 4;
    int m    = lane & 15;
    int Kl   = (hi << 4) + e;
    int kap  = t * 32 + Kl;                  // 0..575
    int kk   = kap >> 6;                     // 0..8
    int c    = kap & 63;
    int o    = (ot << 4) + m;
    wA[tid] = (_Float16)w[(o * 64 + c) * 9 + kk];
}

// ------------- Kernel P2: transpose x -> f16 (B, HW, C) so gathers are
// contiguous 32-byte rows instead of 16KB-strided scalar loads + cvt chains.
__global__ __launch_bounds__(256) void xprep_kernel(
    const float* __restrict__ x, _Float16* __restrict__ xT)
{
    int g = blockIdx.x * 256 + threadIdx.x;   // 0..16383 = b*4096+s
    int b = g >> 12;
    int s = g & 4095;
    const float* xp = x + ((size_t)b << 18) + s;
    v16h rows[4];
#pragma unroll
    for (int v = 0; v < 4; ++v)
#pragma unroll
        for (int e = 0; e < 16; ++e)
            rows[v][e] = (_Float16)xp[(size_t)((v << 4) + e) << 12];
    v16h* dst = (v16h*)(xT + ((size_t)g << 6));
#pragma unroll
    for (int v = 0; v < 4; ++v) dst[v] = rows[v];
}

// ------------- Kernel C: gather + WMMA GEMM + bias/relu/gamma/residual ------
// Per wave: one (b, 16-wide j tile); 4 accumulators cover all 64 out channels.
// B-fragment layout (32x16 f16): lanes 0-15 hold K=0..15, lanes 16-31 K=16..31
// (this code's order; A is packed the same way): K_local = (lane>>4)*16 + e.
// With kappa = t*32 + K_local: the 9 conv taps each feed two K-steps
// (c = 0..31 then 32..63), and each lane's 16 B elements are 16 contiguous
// f16 values of the transposed x -> one 32B load per fragment.
__global__ __launch_bounds__(32) void conv_gemm_kernel(
    const float* __restrict__ x, const _Float16* __restrict__ xT,
    const int* __restrict__ idx, const _Float16* __restrict__ wA,
    const float* __restrict__ bias, const float* __restrict__ gamma,
    float* __restrict__ y)
{
    int wid  = blockIdx.x;            // 0..1023
    int b    = wid >> 8;
    int j0   = (wid & 255) << 4;
    int lane = threadIdx.x;
    int n    = lane & 15;
    int hi   = lane >> 4;
    int j    = j0 + n;

    float g = gamma[0];
    v8f acc0 = {}, acc1 = {}, acc2 = {}, acc3 = {};
    const int idx_base = ((b << 12) + j) * 9;
    const v16h* waBase = ((const v16h*)wA) + lane;   // + (t*4+ot)*32

    for (int kk = 0; kk < 9; ++kk) {
        int s = idx[idx_base + kk];
        // row of transposed x for source position s: 64 contiguous halfs
        const v16h* xp = (const v16h*)(xT + ((((size_t)(b << 12)) + s) << 6));
        v16h b0 = xp[hi];        // step t=2kk   : c = hi*16 .. hi*16+15
        v16h b1 = xp[2 + hi];    // step t=2kk+1 : c = 32+hi*16 .. +15

        const v16h* wa0 = waBase + ((2 * kk) << 7);       // (t*4)*32
        v16h a00 = wa0[0],  a01 = wa0[32], a02 = wa0[64], a03 = wa0[96];
        const v16h* wa1 = wa0 + 128;
        v16h a10 = wa1[0],  a11 = wa1[32], a12 = wa1[64], a13 = wa1[96];

        acc0 = __builtin_amdgcn_wmma_f32_16x16x32_f16(false, a00, false, b0, (short)0, acc0, false, false);
        acc1 = __builtin_amdgcn_wmma_f32_16x16x32_f16(false, a01, false, b0, (short)0, acc1, false, false);
        acc2 = __builtin_amdgcn_wmma_f32_16x16x32_f16(false, a02, false, b0, (short)0, acc2, false, false);
        acc3 = __builtin_amdgcn_wmma_f32_16x16x32_f16(false, a03, false, b0, (short)0, acc3, false, false);
        acc0 = __builtin_amdgcn_wmma_f32_16x16x32_f16(false, a10, false, b1, (short)0, acc0, false, false);
        acc1 = __builtin_amdgcn_wmma_f32_16x16x32_f16(false, a11, false, b1, (short)0, acc1, false, false);
        acc2 = __builtin_amdgcn_wmma_f32_16x16x32_f16(false, a12, false, b1, (short)0, acc2, false, false);
        acc3 = __builtin_amdgcn_wmma_f32_16x16x32_f16(false, a13, false, b1, (short)0, acc3, false, false);
    }

    // C/D layout: VGPR r -> M = r + hi*8, N = lane&15
#pragma unroll
    for (int r = 0; r < 8; ++r) {
        int m = r + (hi << 3);
        float vals[4] = { acc0[r], acc1[r], acc2[r], acc3[r] };
#pragma unroll
        for (int ot = 0; ot < 4; ++ot) {
            int o = (ot << 4) + m;
            float v = vals[ot] + bias[o];
            v = v > 0.f ? v : 0.f;
            size_t off = (((size_t)((b << 6) + o)) << 12) + j;
            y[off] = fmaf(g, v, x[off]);
        }
    }
}

// ---------------------------- launcher --------------------------------------
extern "C" void kernel_launch(void* const* d_in, const int* in_sizes, int n_in,
                              void* d_out, int out_size, void* d_ws, size_t ws_size,
                              hipStream_t stream)
{
    const float* x     = (const float*)d_in[0];
    const float* qw    = (const float*)d_in[1];
    const float* qb    = (const float*)d_in[2];
    const float* kw    = (const float*)d_in[3];
    const float* kb    = (const float*)d_in[4];
    const float* gamma = (const float*)d_in[5];
    const float* w     = (const float*)d_in[6];
    const float* bias  = (const float*)d_in[7];
    float* y = (float*)d_out;

    char* ws = (char*)d_ws;
    float*    qo   = (float*)(ws);                              // 131072 f32  @ 0
    float*    ko   = (float*)(ws + 512 * 1024);                 // 131072 f32  @ 512K
    int*      idxo = (int*)  (ws + 1024 * 1024);                // 147456 i32  @ 1M
    _Float16* wA   = (_Float16*)(ws + 1638400);                 // 36864 f16   (32B aligned)
    _Float16* xT   = (_Float16*)(ws + 1712128);                 // 1048576 f16 (32B aligned)

    qk_proj_kernel<<<64, 256, 0, stream>>>(x, qw, qb, kw, kb, qo, ko);
    wprep_kernel<<<144, 256, 0, stream>>>(w, wA);
    xprep_kernel<<<64, 256, 0, stream>>>(x, xT);
    topk9_kernel<<<16384, 32, 0, stream>>>(qo, ko, idxo);
    conv_gemm_kernel<<<1024, 32, 0, stream>>>(x, xT, idxo, wA, bias, gamma, y);
}
